// AttenConv_38130719654350
// MI455X (gfx1250) — matrix operand order repaired and checked by reading, not compiled, not run on hardware
//
#include <hip/hip_runtime.h>
#include <hip/hip_bf16.h>
#include <math.h>

typedef __attribute__((ext_vector_type(16))) _Float16 v16h;
typedef __attribute__((ext_vector_type(8)))  _Float16 v8h;
typedef __attribute__((ext_vector_type(8)))  float    v8f;

#define D_DIM   128
#define N_ROWS  8192
#define NNZ_E   262144
#define KSPLIT  2
#define KEYS_PER_SPLIT (N_ROWS / KSPLIT)

static __device__ __forceinline__ v8f wmma_f16(v16h a, v16h b, v8f c) {
    return __builtin_amdgcn_wmma_f32_16x16x32_f16(false, a, false, b, (short)0, c, false, false);
}

// ---------------------------------------------------------------------------
// Wt[n*128+k] = (f16) W[k*128+n]   (so B-operand column loads are contiguous)
// ---------------------------------------------------------------------------
__global__ void prep_wt_kernel(const float* __restrict__ W, _Float16* __restrict__ Wt) {
    int t = blockIdx.x * blockDim.x + threadIdx.x;      // 16384 threads
    int n = t >> 7, k = t & 127;
    Wt[n * D_DIM + k] = (_Float16)W[k * D_DIM + n];
}

// ---------------------------------------------------------------------------
// Y[m,:] = X[m,:] @ W  for 128x128 W (given as Wt f16, transposed row-major).
// One wave -> 16 rows x 128 cols of output. K = 128 = 4 chunks of 32.
// All 8 B-operands of a k-chunk are preloaded in one clause so WMMAs run
// with staggered (non-zero) loadcnt waits instead of load->wait0->wmma.
// ---------------------------------------------------------------------------
__global__ void gemm_rowsW_kernel(const float* __restrict__ X,
                                  const _Float16* __restrict__ Wt,
                                  float* __restrict__ Y, int nrows) {
    int wave = (blockIdx.x * blockDim.x + threadIdx.x) >> 5;
    int lane = threadIdx.x & 31;
    int m0 = wave * 16;
    if (m0 >= nrows) return;
    int ln = lane & 15;          // A: row m ; B: col n ; C: col n
    int h  = lane >> 4;

    // A operands: row (m0+ln), f32 -> f16
    v16h A[4];
    const float* xrow = X + (size_t)(m0 + ln) * D_DIM;
#pragma unroll
    for (int kc = 0; kc < 4; ++kc) {
#pragma unroll
        for (int j = 0; j < 8; ++j) {
            A[kc][j]     = (_Float16)xrow[kc * 32 + h * 16 + j];
            A[kc][8 + j] = (_Float16)xrow[kc * 32 + h * 16 + 8 + j];
        }
    }

    v8f C[8];
#pragma unroll
    for (int nt = 0; nt < 8; ++nt) C[nt] = (v8f){};

#pragma unroll
    for (int kc = 0; kc < 4; ++kc) {
        v16h B[8];
#pragma unroll
        for (int nt = 0; nt < 8; ++nt)
            B[nt] = *(const v16h*)(Wt + (size_t)(nt * 16 + ln) * D_DIM + kc * 32 + h * 16);
#pragma unroll
        for (int nt = 0; nt < 8; ++nt)
            C[nt] = wmma_f16(A[kc], B[nt], C[nt]);
    }

#pragma unroll
    for (int nt = 0; nt < 8; ++nt)
#pragma unroll
        for (int r = 0; r < 8; ++r)
            Y[(size_t)(m0 + r + 8 * h) * D_DIM + nt * 16 + ln] = C[nt][r];
}

// ---------------------------------------------------------------------------
__global__ void zero_f32_kernel(float* __restrict__ p, int n) {
    int t = blockIdx.x * blockDim.x + threadIdx.x;
    if (t < n) p[t] = 0.0f;
}

// one thread per (edge, dim): both scatters at once (global_atomic_add_f32)
__global__ void scatter_edges_kernel(const float* __restrict__ val,
                                     const int* __restrict__ row,
                                     const int* __restrict__ col,
                                     const float* __restrict__ ej,
                                     const float* __restrict__ ek,
                                     float* __restrict__ un,
                                     float* __restrict__ in_) {
    long t = (long)blockIdx.x * blockDim.x + threadIdx.x;   // NNZ*128 threads
    long e = t >> 7;
    int  d = (int)(t & 127);
    if (e >= NNZ_E) return;
    float v = val[e];
    int r = row[e], c = col[e];
    atomicAdd(un  + (size_t)r * D_DIM + d, v * ek[(size_t)c * D_DIM + d]);
    atomicAdd(in_ + (size_t)c * D_DIM + d, v * ej[(size_t)r * D_DIM + d]);
}

// f32 -> f16 for Q, K; V transposed to [D x I] so PV B-operand loads are contiguous
__global__ void convert_qkv_kernel(const float* __restrict__ un,
                                   const float* __restrict__ in_,
                                   const float* __restrict__ ek,
                                   _Float16* __restrict__ qh,
                                   _Float16* __restrict__ kh,
                                   _Float16* __restrict__ vT) {
    int t = blockIdx.x * blockDim.x + threadIdx.x;          // 8192*128 threads
    int i = t >> 7, d = t & 127;
    qh[t] = (_Float16)un[t];
    kh[t] = (_Float16)in_[t];
    vT[(size_t)d * N_ROWS + i] = (_Float16)ek[t];
}

// ---------------------------------------------------------------------------
// Flash attention, split-K over the key dimension. One single-wave workgroup
// handles 16 query rows x KEYS_PER_SPLIT keys and writes unnormalized partial
// O plus per-row (m, l) stats. Single-wave workgroup => barrier ops are S_NOP,
// so the LDS P-transpose sync is free.
// ---------------------------------------------------------------------------
__global__ void __launch_bounds__(32)
flash_attn_split_kernel(const _Float16* __restrict__ qh,
                        const _Float16* __restrict__ kh,
                        const _Float16* __restrict__ vT,
                        float* __restrict__ Opart,      // [KSPLIT*N_ROWS, 128]
                        float* __restrict__ mstat,      // [KSPLIT*N_ROWS]
                        float* __restrict__ lstat) {    // [KSPLIT*N_ROWS]
    __shared__ __align__(16) _Float16 plds[16 * 32];    // P transpose slice
    int lane = threadIdx.x & 31;
    int tile = blockIdx.x >> 1;                         // M tile index
    int spl  = blockIdx.x & 1;                          // key-split index
    int m0   = tile * 16;
    int k0   = spl * KEYS_PER_SPLIT;
    int ln = lane & 15;
    int h  = lane >> 4;

    // Q tile in registers: 16 x 128 f16 as 4 A-operands
    v16h Q[4];
    const _Float16* qrow = qh + (size_t)(m0 + ln) * D_DIM;
#pragma unroll
    for (int kc = 0; kc < 4; ++kc) {
        v8h lo = *(const v8h*)(qrow + kc * 32 + h * 16);
        v8h hi = *(const v8h*)(qrow + kc * 32 + h * 16 + 8);
#pragma unroll
        for (int j = 0; j < 8; ++j) { Q[kc][j] = lo[j]; Q[kc][8 + j] = hi[j]; }
    }

    v8f O[8];
#pragma unroll
    for (int nt = 0; nt < 8; ++nt) O[nt] = (v8f){};
    float mrow[8], lrow[8];
#pragma unroll
    for (int r = 0; r < 8; ++r) { mrow[r] = -1e30f; lrow[r] = 0.0f; }

    for (int kt = 0; kt < KEYS_PER_SPLIT / 32; ++kt) {
        const _Float16* kr0 = kh + (size_t)(k0 + kt * 32 + ln) * D_DIM;
        const _Float16* kr1 = kh + (size_t)(k0 + kt * 32 + 16 + ln) * D_DIM;
        // prefetch next key tile rows (global_prefetch_b8; L2-resident anyway)
        __builtin_prefetch(kr0 + 32 * D_DIM, 0, 3);
        __builtin_prefetch(kr1 + 32 * D_DIM, 0, 3);

        // ---- preload all K B-operands, then S = Q @ K^T ----
        v16h KB[8];
#pragma unroll
        for (int kc = 0; kc < 4; ++kc) {
            KB[kc]     = *(const v16h*)(kr0 + kc * 32 + h * 16);
            KB[4 + kc] = *(const v16h*)(kr1 + kc * 32 + h * 16);
        }
        v8f S0 = {}, S1 = {};
#pragma unroll
        for (int kc = 0; kc < 4; ++kc) {
            S0 = wmma_f16(Q[kc], KB[kc],     S0);
            S1 = wmma_f16(Q[kc], KB[4 + kc], S1);
        }

        // ---- preload all V B-operands (overlaps the softmax VALU chain) ----
        v16h Bv[8];
#pragma unroll
        for (int nt = 0; nt < 8; ++nt)
            Bv[nt] = *(const v16h*)(vT + (size_t)(nt * 16 + ln) * N_ROWS +
                                    k0 + kt * 32 + h * 16);

        // ---- online softmax; rows live in 16-lane halves (xor 1,2,4,8) ----
        float p0[8], p1[8];
#pragma unroll
        for (int r = 0; r < 8; ++r) {
            float mx = fmaxf(S0[r], S1[r]);
            mx = fmaxf(mx, __shfl_xor(mx, 1));
            mx = fmaxf(mx, __shfl_xor(mx, 2));
            mx = fmaxf(mx, __shfl_xor(mx, 4));
            mx = fmaxf(mx, __shfl_xor(mx, 8));
            float mnew = fmaxf(mrow[r], mx);
            float sc   = __expf(mrow[r] - mnew);
            mrow[r] = mnew;
            float e0 = __expf(S0[r] - mnew);
            float e1 = __expf(S1[r] - mnew);
            float ls = e0 + e1;
            ls += __shfl_xor(ls, 1);
            ls += __shfl_xor(ls, 2);
            ls += __shfl_xor(ls, 4);
            ls += __shfl_xor(ls, 8);
            lrow[r] = lrow[r] * sc + ls;
            p0[r] = e0; p1[r] = e1;
#pragma unroll
            for (int nt = 0; nt < 8; ++nt) O[nt][r] *= sc;
        }

        // ---- transpose P (C-layout -> A-layout) through LDS; single-wave
        //      workgroup => these barriers cost nothing in hardware ----
        __syncthreads();
#pragma unroll
        for (int r = 0; r < 8; ++r) {
            plds[(r + 8 * h) * 32 + ln]      = (_Float16)p0[r];
            plds[(r + 8 * h) * 32 + 16 + ln] = (_Float16)p1[r];
        }
        __syncthreads();
        v16h P;
        {
            v8h lo = *(const v8h*)(plds + ln * 32 + h * 16);
            v8h hi = *(const v8h*)(plds + ln * 32 + h * 16 + 8);
#pragma unroll
            for (int j = 0; j < 8; ++j) { P[j] = lo[j]; P[8 + j] = hi[j]; }
        }
        __syncthreads();

        // ---- O += P @ V (operands already resident) ----
#pragma unroll
        for (int nt = 0; nt < 8; ++nt)
            O[nt] = wmma_f16(P, Bv[nt], O[nt]);
    }

    // ---- write unnormalized partials + stats ----
    size_t base = (size_t)(spl * N_ROWS + m0);
#pragma unroll
    for (int r = 0; r < 8; ++r) {
#pragma unroll
        for (int nt = 0; nt < 8; ++nt)
            Opart[(base + r + 8 * h) * D_DIM + nt * 16 + ln] = O[nt][r];
    }
    if (ln == 0) {
#pragma unroll
        for (int r = 0; r < 8; ++r) {
            mstat[base + r + 8 * h] = mrow[r];
            lstat[base + r + 8 * h] = lrow[r];
        }
    }
}

// ---------------------------------------------------------------------------
// Merge the KSPLIT partials: acc = sum_s exp(m_s - m) O_s / sum_s exp(m_s - m) l_s
// ---------------------------------------------------------------------------
__global__ void merge_split_kernel(const float* __restrict__ Opart,
                                   const float* __restrict__ mstat,
                                   const float* __restrict__ lstat,
                                   float* __restrict__ acc) {
    int t = blockIdx.x * blockDim.x + threadIdx.x;      // N_ROWS*128 threads
    int row = t >> 7;
    float m0 = mstat[row], m1 = mstat[N_ROWS + row];
    float l0 = lstat[row], l1 = lstat[N_ROWS + row];
    float mf = fmaxf(m0, m1);
    float a0 = __expf(m0 - mf), a1 = __expf(m1 - mf);
    float num = Opart[t] * a0 + Opart[(size_t)N_ROWS * D_DIM + t] * a1;
    float den = l0 * a0 + l1 * a1;
    acc[t] = num / den;
}

// ---------------------------------------------------------------------------
extern "C" void kernel_launch(void* const* d_in, const int* in_sizes, int n_in,
                              void* d_out, int out_size, void* d_ws, size_t ws_size,
                              hipStream_t stream) {
    const float* user_emb = (const float*)d_in[0];   // [8192,128]
    const float* item_emb = (const float*)d_in[1];   // [8192,128]
    const float* W        = (const float*)d_in[2];   // [128,128]
    const float* adj_val  = (const float*)d_in[3];   // [262144]
    const int*   adj_row  = (const int*)d_in[4];
    const int*   adj_col  = (const int*)d_in[5];
    float* out = (float*)d_out;                      // [8192,128]

    const size_t MAT = (size_t)N_ROWS * D_DIM;       // 1048576 elements
    float* e_j   = (float*)d_ws;
    float* e_k   = e_j + MAT;
    float* u_n   = e_k + MAT;
    float* i_n   = u_n + MAT;                        // u_n,i_n contiguous
    float* accb  = i_n + MAT;
    float* Opart = accb + MAT;                       // KSPLIT*MAT floats
    float* mstat = Opart + (size_t)KSPLIT * MAT;     // KSPLIT*N_ROWS
    float* lstat = mstat + (size_t)KSPLIT * N_ROWS;
    _Float16* Wt = (_Float16*)(lstat + (size_t)KSPLIT * N_ROWS);
    _Float16* qh = Wt + D_DIM * D_DIM;
    _Float16* kh = qh + MAT;
    _Float16* vT = kh + MAT;

    // 1) Wt = f16(W^T)
    prep_wt_kernel<<<64, 256, 0, stream>>>(W, Wt);
    // 2) projections e_j, e_k
    gemm_rowsW_kernel<<<64, 256, 0, stream>>>(user_emb, Wt, e_j, N_ROWS);
    gemm_rowsW_kernel<<<64, 256, 0, stream>>>(item_emb, Wt, e_k, N_ROWS);
    // 3) zero both neighbor accumulators (contiguous 2*MAT floats), scatter
    zero_f32_kernel<<<(int)((2 * MAT + 255) / 256), 256, 0, stream>>>(u_n, (int)(2 * MAT));
    scatter_edges_kernel<<<(NNZ_E * D_DIM) / 256, 256, 0, stream>>>(
        adj_val, adj_row, adj_col, e_j, e_k, u_n, i_n);
    // 4) f16 conversions + V transpose
    convert_qkv_kernel<<<(int)(MAT / 256), 256, 0, stream>>>(u_n, i_n, e_k, qh, kh, vT);
    // 5) fused flash attention, split-K=2: 1024 single-wave workgroups
    flash_attn_split_kernel<<<(N_ROWS / 16) * KSPLIT, 32, 0, stream>>>(
        qh, kh, vT, Opart, mstat, lstat);
    // 6) merge partials into acc
    merge_split_kernel<<<(int)(MAT / 256), 256, 0, stream>>>(Opart, mstat, lstat, accb);
    // 7) out = acc @ W
    gemm_rowsW_kernel<<<64, 256, 0, stream>>>(accb, Wt, out, N_ROWS);
}
